// SelfAttention_51745765983010
// MI455X (gfx1250) — matrix operand order repaired and checked by reading, not compiled, not run on hardware
//
#include <hip/hip_runtime.h>
#include <hip/hip_bf16.h>
#include <math.h>

// ---------------------------------------------------------------------------
// SelfAttention (masked cross-attention, BATCH=8192, 16 predators x 64
// searchers, dim 256).  HBM-bound (~845 MB/call -> ~36us floor @ 23.3 TB/s),
// so: single fused kernel per batch element, bf16 WMMA (v_wmma_f32_16x16x32_bf16)
// for all matmuls, algebraic fusion  scores = (Q Wk) Xs^T + Q.bk  and
// attn = (W Xs) Wv^T + rowsum(W) bv  to avoid materializing K/V.
// ---------------------------------------------------------------------------

typedef __bf16 bf16;
typedef __attribute__((ext_vector_type(16))) __bf16 v16bf;
typedef __attribute__((ext_vector_type(8)))  float  v8f;

#define NP   16
#define NS   64
#define DIM  256

// ---- WMMA fragment loaders (wave32; A and B both use contiguous k = 16*half + e) ----------

// A: 16x32 bf16, row-major source [m][k] with leading dim lda.
// lane m = lane&15, half = lane>>4; elems 0..15 -> k0+half*16+e:
// one contiguous 32B chunk per lane.
__device__ __forceinline__ v16bf load_a_rm(const bf16* A, int lda, int m0,
                                           int k0, int lane) {
  int m = m0 + (lane & 15);
  int half = lane >> 4;
  const bf16* p = A + m * lda + k0 + half * 16;
  v16bf a;
#pragma unroll
  for (int i = 0; i < 16; ++i) a[i] = p[i];
  return a;
}

// B: 32x16 bf16 where B[k][n] is stored row-major as [n][k] (i.e. we hold the
// transpose of the operand, e.g. W for X@W^T).  lane n = lane&15,
// k = half*16 + e: one contiguous 32B chunk per lane.
__device__ __forceinline__ v16bf load_b_nm(const bf16* B, int ldb, int n0,
                                           int k0, int lane) {
  int n = n0 + (lane & 15);
  int half = lane >> 4;
  const bf16* p = B + n * ldb + k0 + half * 16;
  v16bf b;
#pragma unroll
  for (int i = 0; i < 16; ++i) b[i] = p[i];
  return b;
}

// B: 32x16 bf16 where B[k][n] is stored row-major as [k][n] (k-major),
// used for weights @ Xs where Xs itself is the B operand.
__device__ __forceinline__ v16bf load_b_km(const bf16* B, int ldb, int n0,
                                           int k0, int lane) {
  int n = n0 + (lane & 15);
  int half = lane >> 4;
  const bf16* p = B + (k0 + half * 16) * ldb + n;
  v16bf b;
#pragma unroll
  for (int i = 0; i < 16; ++i) b[i] = p[i * ldb];
  return b;
}

__device__ __forceinline__ v8f wmma_bf16(v16bf a, v16bf b, v8f c) {
  // (neg_a, A, neg_b, B, c_mod, C, reuse_a, reuse_b)
  return __builtin_amdgcn_wmma_f32_16x16x32_bf16(false, a, false, b,
                                                 (short)0, c, false, false);
}

// ---- weight pre-conversion: fp32 -> bf16 into d_ws (512 KB) ---------------
// layout in ws: [Wq | Wk^T | Wv | Wfc], each 256*256 bf16.
__global__ __launch_bounds__(256) void convert_weights(
    const float* __restrict__ Wq, const float* __restrict__ Wk,
    const float* __restrict__ Wv, const float* __restrict__ Wfc,
    bf16* __restrict__ ws) {
  int idx = blockIdx.x * 256 + threadIdx.x;   // 0 .. 4*65536-1
  int m = idx >> 16;
  int e = idx & 65535;
  if (m == 0) {
    ws[e] = (bf16)Wq[e];
  } else if (m == 1) {                         // transpose Wk -> [j][h]
    int h = e >> 8, j = e & 255;
    ws[65536 + j * 256 + h] = (bf16)Wk[e];
  } else if (m == 2) {
    ws[131072 + e] = (bf16)Wv[e];
  } else {
    ws[196608 + e] = (bf16)Wfc[e];
  }
}

// ---- fused attention kernel: one block (8 wave32) per batch element -------
__global__ __launch_bounds__(256) void attn_fused(
    const float* __restrict__ obs, const float* __restrict__ pfg,
    const float* __restrict__ sfg, const bf16* __restrict__ wsb,
    const float* __restrict__ bq, const float* __restrict__ bk,
    const float* __restrict__ bv, const float* __restrict__ bfc,
    float* __restrict__ out, int batch) {
  // LDS pool, phase-aliased regions; total 63,616 B.
  __shared__ __align__(16) unsigned char pool[63616];
  bf16*  Xs      = (bf16*)(pool);            // [64][256] searcher feats (all phases)
  bf16*  Xp      = (bf16*)(pool + 32768);    // [16][256] predator feats (ph1-2)
  float* sw      = (float*)(pool + 32768);   // [16][64] scores/weights (ph4+) aliases Xp
  bf16*  Qs      = (bf16*)(pool + 40960);    // [16][256] Q (ph2-3)
  bf16*  W2s     = (bf16*)(pool + 40960);    // [16][256] weights@Xs (ph7+) aliases Qs
  bf16*  Ts      = (bf16*)(pool + 49152);    // [16][256] Q@Wk (ph3-4)
  bf16*  Ab      = (bf16*)(pool + 49152);    // [16][256] attn (ph8+) aliases Ts
  bf16*  wb      = (bf16*)(pool + 57344);    // [16][64] weights bf16
  float* maskAdd = (float*)(pool + 59392);   // [16][64] 0 or -inf
  float* Qbk     = (float*)(pool + 63488);   // [16]
  int*   flg     = (int*)(pool + 63552);     // [16] no_searcher flags

  const int tid  = threadIdx.x;
  const int lane = tid & 31;
  const int wave = tid >> 5;
  const int b    = blockIdx.x;
  const float NEG_INF = -__builtin_inff();

  const bf16* Wq_b  = wsb;
  const bf16* WkT_b = wsb + 65536;
  const bf16* Wv_b  = wsb + 131072;
  const bf16* Wfc_b = wsb + 196608;

  // Phase 1: stage features (fp32 -> bf16) + distance mask.
  {
    const float* src = sfg + (size_t)b * (NS * DIM);
    for (int i = tid * 4; i < NS * DIM; i += 1024) {
      float4 v = *reinterpret_cast<const float4*>(src + i);
      Xs[i] = (bf16)v.x; Xs[i + 1] = (bf16)v.y;
      Xs[i + 2] = (bf16)v.z; Xs[i + 3] = (bf16)v.w;
    }
    const float* srcp = pfg + (size_t)b * (NP * DIM);
    for (int i = tid * 4; i < NP * DIM; i += 1024) {
      float4 v = *reinterpret_cast<const float4*>(srcp + i);
      Xp[i] = (bf16)v.x; Xp[i + 1] = (bf16)v.y;
      Xp[i + 2] = (bf16)v.z; Xp[i + 3] = (bf16)v.w;
    }
    const float* ob = obs + (size_t)b * 160;   // (80 agents x 2)
    for (int i = tid; i < NP * NS; i += 256) {
      int p = i >> 6, s = i & 63;
      float dx = ob[p * 2]     - ob[32 + s * 2];
      float dy = ob[p * 2 + 1] - ob[32 + s * 2 + 1];
      maskAdd[i] = (dx * dx + dy * dy > 0.25f) ? NEG_INF : 0.0f;  // dis > 0.5
    }
  }
  __syncthreads();

  // Phase 2: Q = Xp @ Wq^T + bq   (16 tiles of 16x16, K=256)
  for (int nt = wave; nt < 16; nt += 8) {
    int n0 = nt * 16;
    v8f acc = {};
#pragma unroll
    for (int k0 = 0; k0 < DIM; k0 += 32)
      acc = wmma_bf16(load_a_rm(Xp, DIM, 0, k0, lane),
                      load_b_nm(Wq_b, DIM, n0, k0, lane), acc);
    int n = n0 + (lane & 15), half = lane >> 4;
    float bias = bq[n];
#pragma unroll
    for (int r = 0; r < 8; ++r)
      Qs[(r + half * 8) * DIM + n] = (bf16)(acc[r] + bias);
  }
  __syncthreads();

  // Phase 3: T = Q @ Wk  (B = Wk^T, row-major [j][h]);  Qbk = Q . bk
  if (tid < NP) {
    float s = 0.f;
    for (int h = 0; h < DIM; ++h) s += (float)Qs[tid * DIM + h] * bk[h];
    Qbk[tid] = s;
  }
  for (int nt = wave; nt < 16; nt += 8) {
    int n0 = nt * 16;
    v8f acc = {};
#pragma unroll
    for (int k0 = 0; k0 < DIM; k0 += 32)
      acc = wmma_bf16(load_a_rm(Qs, DIM, 0, k0, lane),
                      load_b_nm(WkT_b, DIM, n0, k0, lane), acc);
    int n = n0 + (lane & 15), half = lane >> 4;
#pragma unroll
    for (int r = 0; r < 8; ++r)
      Ts[(r + half * 8) * DIM + n] = (bf16)acc[r];
  }
  __syncthreads();

  // Phase 4: scores = (T @ Xs^T + Qbk) / 16 + mask   (4 tiles, waves 0-3)
  if (wave < 4) {
    int n0 = wave * 16;
    v8f acc = {};
#pragma unroll
    for (int k0 = 0; k0 < DIM; k0 += 32)
      acc = wmma_bf16(load_a_rm(Ts, DIM, 0, k0, lane),
                      load_b_nm(Xs, DIM, n0, k0, lane), acc);
    int n = n0 + (lane & 15), half = lane >> 4;
#pragma unroll
    for (int r = 0; r < 8; ++r) {
      int m = r + half * 8;
      sw[m * NS + n] = (acc[r] + Qbk[m]) * 0.0625f + maskAdd[m * NS + n];
    }
  }
  __syncthreads();

  // Phase 5: row softmax over 64 searchers; all-masked row -> weights 0, flag.
  if (tid < NP) {
    float mx = NEG_INF;
    for (int s = 0; s < NS; ++s) mx = fmaxf(mx, sw[tid * NS + s]);
    if (mx == NEG_INF) {
      for (int s = 0; s < NS; ++s) sw[tid * NS + s] = 0.f;
      flg[tid] = 1;
    } else {
      float sum = 0.f;
      for (int s = 0; s < NS; ++s) {
        float e = __expf(sw[tid * NS + s] - mx);
        sw[tid * NS + s] = e;
        sum += e;
      }
      float inv = 1.f / sum;
      for (int s = 0; s < NS; ++s) sw[tid * NS + s] *= inv;
      flg[tid] = 0;
    }
  }
  __syncthreads();

  // Phase 6: weights -> bf16 A-operand copy.
  for (int i = tid; i < NP * NS; i += 256) wb[i] = (bf16)sw[i];
  __syncthreads();

  // Phase 7: W2 = weights @ Xs   (B is Xs itself, k-major; K=64)
  for (int nt = wave; nt < 16; nt += 8) {
    int n0 = nt * 16;
    v8f acc = {};
#pragma unroll
    for (int k0 = 0; k0 < NS; k0 += 32)
      acc = wmma_bf16(load_a_rm(wb, NS, 0, k0, lane),
                      load_b_km(Xs, DIM, n0, k0, lane), acc);
    int n = n0 + (lane & 15), half = lane >> 4;
#pragma unroll
    for (int r = 0; r < 8; ++r)
      W2s[(r + half * 8) * DIM + n] = (bf16)acc[r];
  }
  __syncthreads();

  // Phase 8: attn = W2 @ Wv^T + rowsum(w)*bv  (rowsum = 1 or 0)
  for (int nt = wave; nt < 16; nt += 8) {
    int n0 = nt * 16;
    v8f acc = {};
#pragma unroll
    for (int k0 = 0; k0 < DIM; k0 += 32)
      acc = wmma_bf16(load_a_rm(W2s, DIM, 0, k0, lane),
                      load_b_nm(Wv_b, DIM, n0, k0, lane), acc);
    int n = n0 + (lane & 15), half = lane >> 4;
    float bvv = bv[n];
#pragma unroll
    for (int r = 0; r < 8; ++r) {
      int m = r + half * 8;
      float sel = flg[m] ? 0.f : 1.f;
      Ab[m * DIM + n] = (bf16)(acc[r] + sel * bvv);
    }
  }
  __syncthreads();

  // Phase 9: out = attn @ Wfc^T + bfc, fallback to pf for no_searcher rows.
  for (int nt = wave; nt < 16; nt += 8) {
    int n0 = nt * 16;
    v8f acc = {};
#pragma unroll
    for (int k0 = 0; k0 < DIM; k0 += 32)
      acc = wmma_bf16(load_a_rm(Ab, DIM, 0, k0, lane),
                      load_b_nm(Wfc_b, DIM, n0, k0, lane), acc);
    int n = n0 + (lane & 15), half = lane >> 4;
    float bias = bfc[n];
#pragma unroll
    for (int r = 0; r < 8; ++r) {
      int m = r + half * 8;
      float v = acc[r] + bias;
      if (flg[m]) v = pfg[(size_t)b * (NP * DIM) + m * DIM + n];
      out[(size_t)b * (NP * DIM) + m * DIM + n] = v;
    }
  }

  // weights output (fp32), second section of d_out.
  float* outw = out + (size_t)batch * NP * DIM;
  for (int i = tid; i < NP * NS; i += 256)
    outw[(size_t)b * (NP * NS) + i] = sw[i];
}

// ---------------------------------------------------------------------------
extern "C" void kernel_launch(void* const* d_in, const int* in_sizes, int n_in,
                              void* d_out, int out_size, void* d_ws,
                              size_t ws_size, hipStream_t stream) {
  (void)n_in; (void)out_size; (void)ws_size;
  const float* obs = (const float*)d_in[0];
  const float* pf  = (const float*)d_in[1];
  const float* sf  = (const float*)d_in[2];
  // d_in[3] = batch_size scalar (device); derive batch from obs size instead.
  const float* Wq  = (const float*)d_in[4];
  const float* bq  = (const float*)d_in[5];
  const float* Wk  = (const float*)d_in[6];
  const float* bk  = (const float*)d_in[7];
  const float* Wv  = (const float*)d_in[8];
  const float* bv  = (const float*)d_in[9];
  const float* Wfc = (const float*)d_in[10];
  const float* bfc = (const float*)d_in[11];

  int batch = in_sizes[0] / 160;            // obs is (B, 80, 2)
  bf16* wsb = (bf16*)d_ws;                  // needs 4*65536*2 = 512 KB

  convert_weights<<<1024, 256, 0, stream>>>(Wq, Wk, Wv, Wfc, wsb);
  attn_fused<<<batch, 256, 0, stream>>>(obs, pf, sf, wsb, bq, bk, bv, bfc,
                                        (float*)d_out, batch);
}
